// WindowAttentionRFB_71863392797252
// MI455X (gfx1250) — compile-verified, matrix operand order checked by reading, not run on hardware
//
#include <hip/hip_runtime.h>

// ---------------------------------------------------------------------------
// WindowAttentionRFB for gfx1250 (MI455X)
//   bias1 (MLP -> f16 bq/bk) -> qkv WMMA GEMM -> fused attention
//   (scores = q k^T + bq bk^T as two chained WMMAs, wave-cooperative softmax,
//    attn@V) -> proj WMMA GEMM.
//   Async global->LDS staging (global_load_async_to_lds_b128) for pure f16
//   copies; f32->f16 converting stages use regular vector loads.
// Shapes: B=512, N=256, C=256, H=8, hd=32, R=32
// ---------------------------------------------------------------------------

typedef __attribute__((ext_vector_type(16))) _Float16 v16h;
typedef __attribute__((ext_vector_type(8)))  float    v8f;

union Frag16 { v16h v; uint4 q[2]; _Float16 h[16]; };
union Acc8   { v8f  v; float f[8]; };

#define WMMA16(a, b, c) __builtin_amdgcn_wmma_f32_16x16x32_f16( \
    false, (a), false, (b), (short)0, (c), false, false)

static __device__ __forceinline__ v16h lds_frag16(const _Float16* rowbase, int hsel) {
  // 16-bit A/B fragment: lane (m = lane&15, hsel = lane>>4)
  // chunk0: K = 8*hsel .. +7  -> VGPR0..3 ; chunk1: K = 16+8*hsel .. +7 -> VGPR4..7
  Frag16 t;
  const _Float16* p = rowbase + hsel * 8;
  t.q[0] = *(const uint4*)(p);
  t.q[1] = *(const uint4*)(p + 16);
  return t.v;
}

// 64B per-lane global -> LDS async copy (4 x b128), tracked on ASYNCcnt.
static __device__ __forceinline__ void async_copy64(void* lds_dst, const void* gsrc) {
  unsigned l = (unsigned)(uintptr_t)lds_dst;        // LDS offset = low 32 bits of flat addr
  unsigned long long g = (unsigned long long)(uintptr_t)gsrc;
  asm volatile(
      "global_load_async_to_lds_b128 %0, %1, off\n\t"
      "global_load_async_to_lds_b128 %0, %1, off offset:16\n\t"
      "global_load_async_to_lds_b128 %0, %1, off offset:32\n\t"
      "global_load_async_to_lds_b128 %0, %1, off offset:48"
      :: "v"(l), "v"(g) : "memory");
}

// 16B per-lane global -> LDS async copy.
static __device__ __forceinline__ void async_copy16(void* lds_dst, const void* gsrc) {
  unsigned l = (unsigned)(uintptr_t)lds_dst;
  unsigned long long g = (unsigned long long)(uintptr_t)gsrc;
  asm volatile("global_load_async_to_lds_b128 %0, %1, off"
               :: "v"(l), "v"(g) : "memory");
}

static __device__ __forceinline__ void wait_async0() {
  asm volatile("s_wait_asynccnt 0x0" ::: "memory");
}

// ---------------------------------------------------------------------------
// Kernel 1: position-bias MLP -> f16 bq [H,N,R] (pre-scaled), bk [H,N,R]
// grid 16 x 256; block bx covers output columns [32*bx, 32*bx+32)
// ---------------------------------------------------------------------------
__global__ void bias1_kernel(const float* __restrict__ w1, const float* __restrict__ b1,
                             const float* __restrict__ w2, const float* __restrict__ bscale_p,
                             _Float16* __restrict__ bq, _Float16* __restrict__ bk) {
  __shared__ float hf[256][129];            // padded: conflict-free column reads
  const int n = threadIdx.x;                // position index 0..255
  const float invden = 1.0f / logf(16.0f);  // log1p(15) = log(16)
  const float c0 = log1pf((float)(n >> 4)) * invden;   // gy
  const float c1 = log1pf((float)(n & 15)) * invden;   // gx
  for (int j = 0; j < 128; ++j) {
    float z = c0 * w1[j] + c1 * w1[128 + j] + b1[j];
    hf[n][j] = 0.5f * z * (1.0f + erff(z * 0.70710678118654752f));  // exact gelu
  }
  __syncthreads();
  const float bsc = bscale_p[0];
  int jj = blockIdx.x * 32;
  for (int e = 0; e < 32; ++e, ++jj) {
    float acc = 0.0f;
    for (int j = 0; j < 128; ++j) acc += hf[n][j] * w2[j * 512 + jj];
    float s = 1.0f / (1.0f + __expf(-acc));
    int h = jj >> 6, which = (jj >> 5) & 1, r = jj & 31;
    if (which == 0) bq[(h * 256 + n) * 32 + r] = (_Float16)(bsc * s);
    else            bk[(h * 256 + n) * 32 + r] = (_Float16)s;
  }
}

// ---------------------------------------------------------------------------
// Kernel 2: QKV GEMM  [131072 x 256] @ [256 x 768] (+bias, q*=logit_scale)
// Block tile 128x128, 8 waves, each wave 2x4 16x16 accum tiles.
// ---------------------------------------------------------------------------
__global__ void qkv_kernel(const float* __restrict__ x, const float* __restrict__ w,
                           const float* __restrict__ bvec, const float* __restrict__ lsc,
                           _Float16* __restrict__ qo, _Float16* __restrict__ ko,
                           _Float16* __restrict__ vo) {
  __shared__ _Float16 As[128][40];   // pitch 40 h (80 B, 16B-aligned)
  __shared__ _Float16 Bs[128][40];   // transposed B tile: Bs[col][k]
  const int tid = threadIdx.x, lane = tid & 31, wave = tid >> 5;
  const int wm = wave & 3, wn = wave >> 2;
  const int mlane = lane & 15, hsel = lane >> 4;
  const int m0 = blockIdx.y * 128, n0 = blockIdx.x * 128;

  Acc8 acc[2][4];
  #pragma unroll
  for (int i = 0; i < 2; ++i)
    #pragma unroll
    for (int j = 0; j < 4; ++j)
      #pragma unroll
      for (int r = 0; r < 8; ++r) acc[i][j].f[r] = 0.0f;

  for (int kk = 0; kk < 256; kk += 32) {
    __syncthreads();
    // prefetch next K chunk into WGP-level cache (locality 3 -> WGP scope)
    if (kk + 32 < 256) {
      __builtin_prefetch(x + (size_t)(m0 + (tid >> 3)) * 256 + kk + 32 + (tid & 7) * 4, 0, 3);
      __builtin_prefetch(w + (size_t)(kk + 32 + (tid >> 5)) * 768 + n0 + (tid & 31) * 4, 0, 3);
    }
    // A: x[m0+r][kk + k4..k4+3], f32 -> f16
    #pragma unroll
    for (int p = 0; p < 4; ++p) {
      int r = (tid >> 3) + p * 32;
      float4 a = *(const float4*)(x + (size_t)(m0 + r) * 256 + kk + (tid & 7) * 4);
      union { _Float16 h[4]; uint2 u; } t;
      t.h[0] = (_Float16)a.x; t.h[1] = (_Float16)a.y;
      t.h[2] = (_Float16)a.z; t.h[3] = (_Float16)a.w;
      *(uint2*)&As[r][(tid & 7) * 4] = t.u;
    }
    // B transposed into LDS: Bs[c][k] = w[(kk+k)*768 + n0 + c]
    #pragma unroll
    for (int p = 0; p < 4; ++p) {
      int k = (tid >> 5) + p * 8;
      int c = (tid & 31) * 4;
      float4 b = *(const float4*)(w + (size_t)(kk + k) * 768 + n0 + c);
      Bs[c + 0][k] = (_Float16)b.x; Bs[c + 1][k] = (_Float16)b.y;
      Bs[c + 2][k] = (_Float16)b.z; Bs[c + 3][k] = (_Float16)b.w;
    }
    __syncthreads();
    Frag16 af[2], bf[4];
    #pragma unroll
    for (int i = 0; i < 2; ++i) af[i].v = lds_frag16(&As[wm * 32 + i * 16 + mlane][0], hsel);
    #pragma unroll
    for (int j = 0; j < 4; ++j) bf[j].v = lds_frag16(&Bs[wn * 64 + j * 16 + mlane][0], hsel);
    #pragma unroll
    for (int i = 0; i < 2; ++i)
      #pragma unroll
      for (int j = 0; j < 4; ++j)
        acc[i][j].v = WMMA16(af[i].v, bf[j].v, acc[i][j].v);
  }

  // epilogue: +bias, q*=logit_scale[h], scatter f16 to [B,H,N,32]
  #pragma unroll
  for (int j = 0; j < 4; ++j) {
    const int jg = n0 + wn * 64 + j * 16 + mlane;     // global out column 0..767
    const int which = jg >> 8, h = (jg >> 5) & 7, d = jg & 31;
    const float addb = bvec[jg];
    const float scale = (which == 0) ? lsc[h] : 1.0f;
    _Float16* outp = (which == 0) ? qo : ((which == 1) ? ko : vo);
    #pragma unroll
    for (int i = 0; i < 2; ++i) {
      #pragma unroll
      for (int r = 0; r < 8; ++r) {
        int mg = m0 + wm * 32 + i * 16 + r + 8 * hsel;  // global row = b*256 + nrow
        int bb = mg >> 8, nrow = mg & 255;
        float val = (acc[i][j].f[r] + addb) * scale;
        outp[(((size_t)bb * 8 + h) * 256 + nrow) * 32 + d] = (_Float16)val;
      }
    }
  }
}

// ---------------------------------------------------------------------------
// Kernel 3: fused attention per (b,h).  grid (8, 512), 256 threads = 8 waves.
//  scores tile = WMMA(q,k^T) chained with WMMA(bq,bk^T)  (bias folded in)
//  q/k/bq/bk staged with async-to-LDS copies; v transposed through VGPRs.
// ---------------------------------------------------------------------------
__global__ void attn_kernel(const _Float16* __restrict__ qg, const _Float16* __restrict__ kg,
                            const _Float16* __restrict__ vg,
                            const _Float16* __restrict__ bqg, const _Float16* __restrict__ bkg,
                            const float* __restrict__ vrs_g, _Float16* __restrict__ ao) {
  __shared__ _Float16 qs[256][40];     // 20 KB
  __shared__ _Float16 ks[256][40];     // 20 KB
  __shared__ _Float16 bqs[256][40];    // 20 KB  (rank-bias q factors, head h)
  __shared__ _Float16 bks[256][40];    // 20 KB  (rank-bias k factors, head h)
  __shared__ _Float16 vt[32][264];     // v transposed: vt[d][m], 16.9 KB
  __shared__ _Float16 sc[256][264];    // scores -> attn weights (f16), 135 KB
  __shared__ float    vmean[32];

  const int h = blockIdx.x, b = blockIdx.y;
  const int tid = threadIdx.x, lane = tid & 31, wave = tid >> 5;
  const int mlane = lane & 15, hsel = lane >> 4;
  const size_t base = ((size_t)b * 8 + h) * 256 * 32;

  // phase 0: stage q, k, bq, bk (async f16 copies) and v (transpose in VGPRs)
  async_copy64(&qs[tid][0],  qg + base + (size_t)tid * 32);
  async_copy64(&ks[tid][0],  kg + base + (size_t)tid * 32);
  async_copy64(&bqs[tid][0], bqg + ((size_t)h * 256 + tid) * 32);
  async_copy64(&bks[tid][0], bkg + ((size_t)h * 256 + tid) * 32);
  {
    union { uint4 u[4]; _Float16 h16[32]; } vv;
    const uint4* vp = (const uint4*)(vg + base + (size_t)tid * 32);
    #pragma unroll
    for (int c = 0; c < 4; ++c) vv.u[c] = vp[c];
    #pragma unroll
    for (int d = 0; d < 32; ++d) vt[d][tid] = vv.h16[d];
  }
  wait_async0();
  __syncthreads();

  // v_mean (wave 0)
  if (tid < 32) {
    float s = 0.0f;
    for (int m = 0; m < 256; ++m) s += (float)vt[tid][m];
    vmean[tid] = s * (1.0f / 256.0f);
  }

  // phase 1: scores = q k^T + bq bk^T   (two chained WMMAs per 16x16 tile)
  #pragma unroll
  for (int qi = 0; qi < 2; ++qi) {
    const int q0 = (wave * 2 + qi) * 16;
    Frag16 aq; aq.v = lds_frag16(&qs[q0 + mlane][0], hsel);
    Frag16 ab; ab.v = lds_frag16(&bqs[q0 + mlane][0], hsel);
    for (int kt = 0; kt < 16; ++kt) {
      const int k0 = kt * 16;
      Frag16 bk_; bk_.v = lds_frag16(&ks[k0 + mlane][0], hsel);
      Frag16 bb;  bb.v  = lds_frag16(&bks[k0 + mlane][0], hsel);
      Acc8 c;
      #pragma unroll
      for (int r = 0; r < 8; ++r) c.f[r] = 0.0f;
      c.v = WMMA16(aq.v, bk_.v, c.v);
      c.v = WMMA16(ab.v, bb.v, c.v);
      #pragma unroll
      for (int r = 0; r < 8; ++r)
        sc[q0 + r + 8 * hsel][k0 + mlane] = (_Float16)c.f[r];
    }
  }

  // phase 2: softmax rows (wave w owns rows 32w..32w+31; wave-local in LDS)
  for (int j = 0; j < 32; ++j) {
    const int rr = wave * 32 + j;
    union { uint4 u; _Float16 h16[8]; } rv;
    rv.u = *(const uint4*)&sc[rr][lane * 8];
    float mx = -3.0e38f;
    #pragma unroll
    for (int e = 0; e < 8; ++e) mx = fmaxf(mx, (float)rv.h16[e]);
    #pragma unroll
    for (int off = 16; off > 0; off >>= 1) mx = fmaxf(mx, __shfl_xor(mx, off, 32));
    float ef[8], sum = 0.0f;
    #pragma unroll
    for (int e = 0; e < 8; ++e) { ef[e] = __expf((float)rv.h16[e] - mx); sum += ef[e]; }
    #pragma unroll
    for (int off = 16; off > 0; off >>= 1) sum += __shfl_xor(sum, off, 32);
    const float inv = 1.0f / sum;
    #pragma unroll
    for (int e = 0; e < 8; ++e) rv.h16[e] = (_Float16)(ef[e] * inv);
    *(uint4*)&sc[rr][lane * 8] = rv.u;
  }
  __syncthreads();   // publish vmean, keep phases ordered

  // phase 3: out = attn @ v  (+ vrs * v_mean), store f16 to [B,N,C]
  const float vrs = vrs_g[h];
  #pragma unroll
  for (int qi = 0; qi < 2; ++qi) {
    const int q0 = (wave * 2 + qi) * 16;
    #pragma unroll
    for (int dt = 0; dt < 2; ++dt) {
      const int d0 = dt * 16;
      Acc8 acc;
      #pragma unroll
      for (int r = 0; r < 8; ++r) acc.f[r] = 0.0f;
      #pragma unroll
      for (int mc = 0; mc < 8; ++mc) {
        Frag16 af; af.v = lds_frag16(&sc[q0 + mlane][mc * 32], hsel);
        Frag16 bf; bf.v = lds_frag16(&vt[d0 + mlane][mc * 32], hsel);
        acc.v = WMMA16(af.v, bf.v, acc.v);
      }
      const float vm = vmean[d0 + mlane] * vrs;
      #pragma unroll
      for (int r = 0; r < 8; ++r) {
        int mg = q0 + r + 8 * hsel;
        float val = acc.f[r] + vm;
        ao[((size_t)b * 256 + mg) * 256 + h * 32 + d0 + mlane] = (_Float16)val;
      }
    }
  }
}

// ---------------------------------------------------------------------------
// Kernel 4: proj GEMM  [131072 x 256](f16) @ [256 x 256] + proj_b -> f32 out
// A tile staged with async-to-LDS (pure f16 copy); B tile converted f32->f16.
// ---------------------------------------------------------------------------
__global__ void proj_kernel(const _Float16* __restrict__ a, const float* __restrict__ w,
                            const float* __restrict__ bvec, float* __restrict__ out) {
  __shared__ _Float16 As[128][40];
  __shared__ _Float16 Bs[128][40];
  const int tid = threadIdx.x, lane = tid & 31, wave = tid >> 5;
  const int wm = wave & 3, wn = wave >> 2;
  const int mlane = lane & 15, hsel = lane >> 4;
  const int m0 = blockIdx.y * 128, n0 = blockIdx.x * 128;

  Acc8 acc[2][4];
  #pragma unroll
  for (int i = 0; i < 2; ++i)
    #pragma unroll
    for (int j = 0; j < 4; ++j)
      #pragma unroll
      for (int r = 0; r < 8; ++r) acc[i][j].f[r] = 0.0f;

  const int ar0 = tid >> 2, ak8 = (tid & 3) * 8;
  void* lds_a0 = &As[ar0][ak8];
  void* lds_a1 = &As[ar0 + 64][ak8];

  for (int kk = 0; kk < 256; kk += 32) {
    __syncthreads();
    if (kk + 32 < 256)
      __builtin_prefetch(w + (size_t)(kk + 32 + (tid >> 5)) * 256 + n0 + (tid & 31) * 4, 0, 3);
    // A tile: async f16 copies (16B per lane per pass)
    async_copy16(lds_a0, a + (size_t)(m0 + ar0) * 256 + kk + ak8);
    async_copy16(lds_a1, a + (size_t)(m0 + ar0 + 64) * 256 + kk + ak8);
    // B transposed into LDS with f32 -> f16 conversion
    #pragma unroll
    for (int p = 0; p < 4; ++p) {
      int k = (tid >> 5) + p * 8;
      int c = (tid & 31) * 4;
      float4 b = *(const float4*)(w + (size_t)(kk + k) * 256 + n0 + c);
      Bs[c + 0][k] = (_Float16)b.x; Bs[c + 1][k] = (_Float16)b.y;
      Bs[c + 2][k] = (_Float16)b.z; Bs[c + 3][k] = (_Float16)b.w;
    }
    wait_async0();
    __syncthreads();
    Frag16 af[2], bf[4];
    #pragma unroll
    for (int i = 0; i < 2; ++i) af[i].v = lds_frag16(&As[wm * 32 + i * 16 + mlane][0], hsel);
    #pragma unroll
    for (int j = 0; j < 4; ++j) bf[j].v = lds_frag16(&Bs[wn * 64 + j * 16 + mlane][0], hsel);
    #pragma unroll
    for (int i = 0; i < 2; ++i)
      #pragma unroll
      for (int j = 0; j < 4; ++j)
        acc[i][j].v = WMMA16(af[i].v, bf[j].v, acc[i][j].v);
  }

  #pragma unroll
  for (int j = 0; j < 4; ++j) {
    const int jg = n0 + wn * 64 + j * 16 + mlane;
    const float addb = bvec[jg];
    #pragma unroll
    for (int i = 0; i < 2; ++i) {
      #pragma unroll
      for (int r = 0; r < 8; ++r) {
        int mg = m0 + wm * 32 + i * 16 + r + 8 * hsel;
        out[(size_t)mg * 256 + jg] = acc[i][j].f[r] + addb;
      }
    }
  }
}

// ---------------------------------------------------------------------------
// Host launcher
// ---------------------------------------------------------------------------
extern "C" void kernel_launch(void* const* d_in, const int* in_sizes, int n_in,
                              void* d_out, int out_size, void* d_ws, size_t ws_size,
                              hipStream_t stream) {
  const float* x       = (const float*)d_in[0];
  const float* qkv_w   = (const float*)d_in[1];
  const float* qkv_b   = (const float*)d_in[2];
  const float* proj_w  = (const float*)d_in[3];
  const float* proj_b  = (const float*)d_in[4];
  const float* lsc     = (const float*)d_in[5];
  const float* vrs     = (const float*)d_in[6];
  const float* mlp1_w  = (const float*)d_in[7];
  const float* mlp1_b  = (const float*)d_in[8];
  const float* mlp2_w  = (const float*)d_in[9];
  const float* bscale  = (const float*)d_in[10];

  char* ws = (char*)d_ws;
  const size_t QSZ = (size_t)512 * 8 * 256 * 32 * sizeof(_Float16);  // 67.1 MB each
  _Float16* bq = (_Float16*)(ws + 0);          // 128 KB (f16, pre-scaled by bias_scale)
  _Float16* bk = (_Float16*)(ws + 131072);     // 128 KB
  _Float16* q  = (_Float16*)(ws + 262144);
  _Float16* k  = (_Float16*)(ws + 262144 + QSZ);
  _Float16* v  = (_Float16*)(ws + 262144 + 2 * QSZ);
  _Float16* ao = (_Float16*)(ws + 262144 + 3 * QSZ);
  float*    out = (float*)d_out;

  bias1_kernel<<<16, 256, 0, stream>>>(mlp1_w, mlp1_b, mlp2_w, bscale, bq, bk);
  qkv_kernel<<<dim3(6, 1024), 256, 0, stream>>>(x, qkv_w, qkv_b, lsc, q, k, v);
  attn_kernel<<<dim3(8, 512), 256, 0, stream>>>(q, k, v, bq, bk, vrs, ao);
  proj_kernel<<<dim3(2, 1024), 256, 0, stream>>>(ao, proj_w, proj_b, out);
}
